// LearnedSparseRouter_83897891160584
// MI455X (gfx1250) — matrix operand order repaired and checked by reading, not compiled, not run on hardware
//
#include <hip/hip_runtime.h>
#include <hip/hip_bf16.h>
#include <math.h>

typedef __attribute__((ext_vector_type(16))) __bf16 v16bf;
typedef __attribute__((ext_vector_type(8)))  __bf16 v8bf;
typedef __attribute__((ext_vector_type(4)))  __bf16 v4bf;
typedef __attribute__((ext_vector_type(8)))  float  v8f;
typedef __attribute__((ext_vector_type(4)))  float  v4f;
typedef __attribute__((ext_vector_type(4)))  int    v4i;

#define BB    16384
#define HD    4096
#define HHD   2048
#define EE    64
#define TOPK  12

// ---------------- async global->LDS support (gfx1250) ----------------
#if defined(__has_builtin)
#  if __has_builtin(__builtin_amdgcn_global_load_async_to_lds_b128)
#    define HAS_ASYNC_LDS 1
#  endif
#endif
#ifndef HAS_ASYNC_LDS
#  define HAS_ASYNC_LDS 0
#endif

#if HAS_ASYNC_LDS && __has_builtin(__builtin_amdgcn_s_wait_asynccnt)
#  define WAIT_ASYNC() __builtin_amdgcn_s_wait_asynccnt(0)
#elif HAS_ASYNC_LDS
#  define WAIT_ASYNC() asm volatile("s_wait_asynccnt 0x0" ::: "memory")
#else
#  define WAIT_ASYNC() ((void)0)
#endif

static __device__ __forceinline__ void cp16_g2l(__bf16* lds_dst, const __bf16* gsrc) {
#if HAS_ASYNC_LDS
  __builtin_amdgcn_global_load_async_to_lds_b128((v4i*)gsrc, (v4i*)lds_dst, 0, 0);
#else
  *reinterpret_cast<v8bf*>(lds_dst) = *reinterpret_cast<const v8bf*>(gsrc);
#endif
}

// ---------------- workspace layout (bytes) ----------------
#define OFF_XB    ((size_t)0)                                   // BB*HD bf16  = 128 MB
#define OFF_W1B   (OFF_XB  + (size_t)BB * HD * 2)               // HHD*HD bf16 = 16 MB
#define OFF_W2B   (OFF_W1B + (size_t)HHD * HD * 2)              // EE*HHD bf16
#define OFF_RB    (OFF_W2B + (size_t)EE * HHD * 2)              // EE*HD bf16
#define OFF_HB    (OFF_RB  + (size_t)EE * HD * 2)               // BB*HHD bf16 = 64 MB
#define OFF_GATES (OFF_HB  + (size_t)BB * HHD * 2)              // BB*EE f32   = 4 MB
#define OFF_ACC   (OFF_GATES + (size_t)BB * EE * 4)             // 65 f32

// ---------------- fragment helpers ----------------
static __device__ __forceinline__ v16bf make_frag(v8bf lo, v8bf hi) {
  v16bf r;
#pragma unroll
  for (int i = 0; i < 8; ++i) { r[i] = lo[i]; r[i + 8] = hi[i]; }
  return r;
}

// A fragment: 16x32 (MxK) tile from row-major [rows, ldk] storage.
static __device__ __forceinline__ v16bf load_a_frag(const __bf16* base, int ldk, int lane, int k0) {
  const int row = lane & 15;
  const int cb  = k0 + ((lane >> 4) << 3);   // 0 or 8
  const __bf16* p = base + (size_t)row * ldk + cb;
  v8bf lo = *reinterpret_cast<const v8bf*>(p);        // K = cb..cb+7
  v8bf hi = *reinterpret_cast<const v8bf*>(p + 16);   // K = cb+16..cb+23
  return make_frag(lo, hi);
}

// B fragment: 32x16 (KxN); source is weight rows [N, ldk] (we compute X @ W^T).
static __device__ __forceinline__ v16bf load_b_frag(const __bf16* base, int ldk, int lane, int k0) {
  const int row = lane & 15;                 // N
  const int kb  = k0 + ((lane >> 4) << 3);   // 0 or 8
  const __bf16* p = base + (size_t)row * ldk + kb;
  v8bf lo = *reinterpret_cast<const v8bf*>(p);        // K = kb..kb+7
  v8bf hi = *reinterpret_cast<const v8bf*>(p + 16);   // K = kb+16..kb+23
  return make_frag(lo, hi);
}

// ---------------- conversion kernels (4-wide) ----------------
__global__ void f32_to_bf16_kernel(const v4f* __restrict__ in, v4bf* __restrict__ out, size_t n4) {
  size_t i = (size_t)blockIdx.x * blockDim.x + threadIdx.x;
  if (i < n4) {
    v4f v = in[i];
    v4bf o;
#pragma unroll
    for (int j = 0; j < 4; ++j) o[j] = (__bf16)v[j];
    out[i] = o;
  }
}

__global__ void mask_router_kernel(const v4f* __restrict__ rw, const v4f* __restrict__ sm,
                                   v4bf* __restrict__ out, size_t n4) {
  size_t i = (size_t)blockIdx.x * blockDim.x + threadIdx.x;
  if (i < n4) {
    v4f r = rw[i], s = sm[i];
    v4bf o;
#pragma unroll
    for (int j = 0; j < 4; ++j) {
      float g = 1.f / (1.f + expf(-s[j]));
      o[j] = (__bf16)(r[j] * g);
    }
    out[i] = o;
  }
}

// ---------------- GEMM1: Hb = relu(X @ W1^T + b1), bf16 out ----------------
// Block tile 128(M) x 128(N), K staged 64 at a time, double-buffered LDS,
// async global->LDS copies. 8 waves; wave w owns rows [16w,16w+16) x all 128 cols.
#define G1_SA 72                 // padded LDS row stride (elements) -> conflict-friendly
#define G1_STAGES (HD / 64)      // 64

__global__ __launch_bounds__(256) void gemm1_relu_kernel(const __bf16* __restrict__ Xb,
                                                         const __bf16* __restrict__ W1b,
                                                         const float* __restrict__ b1,
                                                         __bf16* __restrict__ Hb) {
  __shared__ __bf16 sA[2][128][G1_SA];
  __shared__ __bf16 sB[2][128][G1_SA];

  const int tid  = threadIdx.x;
  const int w    = tid >> 5;
  const int lane = tid & 31;
  const int m0   = (blockIdx.x / (HHD / 128)) * 128;
  const int n0   = (blockIdx.x % (HHD / 128)) * 128;

  // copy geometry: thread t loads row t/2, 32-col half (t&1) of both tiles
  const int crow = tid >> 1;
  const int ccol = (tid & 1) * 32;
  const __bf16* gA = Xb  + (size_t)(m0 + crow) * HD + ccol;
  const __bf16* gB = W1b + (size_t)(n0 + crow) * HD + ccol;

  v8f acc[8] = {};

  // prologue: stage 0 into buffer 0
#pragma unroll
  for (int i = 0; i < 4; ++i) {
    cp16_g2l(&sA[0][crow][ccol + i * 8], gA + i * 8);
    cp16_g2l(&sB[0][crow][ccol + i * 8], gB + i * 8);
  }
  WAIT_ASYNC();
  __syncthreads();

  int buf = 0;
  for (int s = 0; s < G1_STAGES; ++s) {
    if (s + 1 < G1_STAGES) {
      const int k1 = (s + 1) * 64;
#pragma unroll
      for (int i = 0; i < 4; ++i) {
        cp16_g2l(&sA[buf ^ 1][crow][ccol + i * 8], gA + k1 + i * 8);
        cp16_g2l(&sB[buf ^ 1][crow][ccol + i * 8], gB + k1 + i * 8);
      }
    }
#pragma unroll
    for (int kc = 0; kc < 2; ++kc) {
      const __bf16* ap = &sA[buf][w * 16 + (lane & 15)][kc * 32 + ((lane >> 4) << 3)];
      v16bf a = make_frag(*reinterpret_cast<const v8bf*>(ap),
                          *reinterpret_cast<const v8bf*>(ap + 16));
#pragma unroll
      for (int n = 0; n < 8; ++n) {
        const __bf16* bp = &sB[buf][n * 16 + (lane & 15)][kc * 32 + ((lane >> 4) << 3)];
        v16bf b = make_frag(*reinterpret_cast<const v8bf*>(bp),
                            *reinterpret_cast<const v8bf*>(bp + 16));
        acc[n] = __builtin_amdgcn_wmma_f32_16x16x32_bf16(false, a, false, b, (short)0, acc[n], false, false);
      }
    }
    WAIT_ASYNC();
    __syncthreads();
    buf ^= 1;
  }

  const int rbase = m0 + w * 16 + ((lane >> 4) << 3);
  const int c0    = n0 + (lane & 15);
#pragma unroll
  for (int n = 0; n < 8; ++n) {
    const int col = c0 + n * 16;
    const float bias = b1[col];
#pragma unroll
    for (int r = 0; r < 8; ++r) {
      float v = acc[n][r] + bias;
      v = v > 0.f ? v : 0.f;
      Hb[(size_t)(rbase + r) * HHD + col] = (__bf16)v;
    }
  }
}

// ---------------- GEMM2: gates = sigmoid(Hb @ W2^T + b2), f32 out ----------------
__global__ __launch_bounds__(256) void gemm2_sigmoid_kernel(const __bf16* __restrict__ Hb,
                                                            const __bf16* __restrict__ W2b,
                                                            const float* __restrict__ b2,
                                                            float* __restrict__ gates) {
  const int mt   = (blockIdx.x << 3) + (threadIdx.x >> 5);  // one wave per 16 rows
  const int lane = threadIdx.x & 31;
  const __bf16* Ab = Hb + (size_t)(mt * 16) * HHD;
  v8f acc[4] = {};
  for (int k0 = 0; k0 < HHD; k0 += 32) {
    v16bf a = load_a_frag(Ab, HHD, lane, k0);
#pragma unroll
    for (int n = 0; n < 4; ++n) {
      const __bf16* Bb = W2b + (size_t)(n * 16) * HHD;
      v16bf b = load_b_frag(Bb, HHD, lane, k0);
      acc[n] = __builtin_amdgcn_wmma_f32_16x16x32_bf16(false, a, false, b, (short)0, acc[n], false, false);
    }
  }
  const int rbase = mt * 16 + ((lane >> 4) << 3);
  const int c0    = lane & 15;
#pragma unroll
  for (int n = 0; n < 4; ++n) {
    const int col = c0 + n * 16;
    const float bias = b2[col];
#pragma unroll
    for (int r = 0; r < 8; ++r) {
      float v = acc[n][r] + bias;
      gates[(size_t)(rbase + r) * EE + col] = 1.f / (1.f + expf(-v));
    }
  }
}

// ---------------- router: scaled = (X @ Rb^T) * gates, f32 out to d_out ----------------
__global__ __launch_bounds__(256) void router_kernel(const __bf16* __restrict__ Xb,
                                                     const __bf16* __restrict__ Rb,
                                                     const float* __restrict__ gates,
                                                     float* __restrict__ scaled) {
  const int mt   = (blockIdx.x << 3) + (threadIdx.x >> 5);
  const int lane = threadIdx.x & 31;
  const __bf16* Ab = Xb + (size_t)(mt * 16) * HD;
  v8f acc[4] = {};
  for (int k0 = 0; k0 < HD; k0 += 32) {
    v16bf a = load_a_frag(Ab, HD, lane, k0);
#pragma unroll
    for (int n = 0; n < 4; ++n) {
      const __bf16* Bb = Rb + (size_t)(n * 16) * HD;
      v16bf b = load_b_frag(Bb, HD, lane, k0);
      acc[n] = __builtin_amdgcn_wmma_f32_16x16x32_bf16(false, a, false, b, (short)0, acc[n], false, false);
    }
  }
  const int rbase = mt * 16 + ((lane >> 4) << 3);
  const int c0    = lane & 15;
#pragma unroll
  for (int n = 0; n < 4; ++n) {
    const int col = c0 + n * 16;
#pragma unroll
    for (int r = 0; r < 8; ++r) {
      const size_t o = (size_t)(rbase + r) * EE + col;
      scaled[o] = acc[n][r] * gates[o];   // TEMPERATURE == 1
    }
  }
}

// ---------------- top-k + dual softmax + load stats ----------------
__global__ void zero_accum_kernel(float* accum) {
  if (threadIdx.x < EE + 1) accum[threadIdx.x] = 0.f;
}

__global__ __launch_bounds__(256) void finalize_kernel(const float* __restrict__ scaled,
                                                       float* __restrict__ out_idx,
                                                       float* __restrict__ out_w,
                                                       float* __restrict__ accum) {
  __shared__ float s_tv[8][TOPK];
  __shared__ int   s_ti[8][TOPK];
  __shared__ float s_load[EE];
  __shared__ float s_ent;
  const int tid  = threadIdx.x;
  const int w    = tid >> 5;
  const int lane = tid & 31;
  if (tid < EE) s_load[tid] = 0.f;
  if (tid == 0) s_ent = 0.f;
  __syncthreads();

  const int row = blockIdx.x * 8 + w;
  const float v0 = scaled[(size_t)row * EE + lane];
  const float v1 = scaled[(size_t)row * EE + 32 + lane];
  float w0 = v0, w1 = v1;

  for (int t = 0; t < TOPK; ++t) {
    float bv = w0; int bi = lane;
    if (w1 > bv) { bv = w1; bi = lane + 32; }
#pragma unroll
    for (int off = 16; off > 0; off >>= 1) {
      float ov = __shfl_xor(bv, off, 32);
      int   oi = __shfl_xor(bi, off, 32);
      if (ov > bv || (ov == bv && oi < bi)) { bv = ov; bi = oi; }
    }
    if (lane == 0) { s_tv[w][t] = bv; s_ti[w][t] = bi; }
    if (bi == lane)      w0 = -__builtin_inff();
    if (bi == lane + 32) w1 = -__builtin_inff();
  }
  __syncthreads();

  const float m = s_tv[w][0];            // global row max
  float ek = (lane < TOPK) ? expf(s_tv[w][lane] - m) : 0.f;
  float esum = ek;
#pragma unroll
  for (int off = 16; off > 0; off >>= 1) esum += __shfl_xor(esum, off, 32);
  if (lane < TOPK) {
    out_idx[(size_t)row * TOPK + lane] = (float)s_ti[w][lane];
    out_w[(size_t)row * TOPK + lane]   = ek / esum;
  }

  float p0 = expf(v0 - m), p1 = expf(v1 - m);
  float psum = p0 + p1;
#pragma unroll
  for (int off = 16; off > 0; off >>= 1) psum += __shfl_xor(psum, off, 32);
  p0 /= psum; p1 /= psum;
  float ent = -(p0 * logf(p0 + 1e-8f) + p1 * logf(p1 + 1e-8f));
#pragma unroll
  for (int off = 16; off > 0; off >>= 1) ent += __shfl_xor(ent, off, 32);

  atomicAdd(&s_load[lane], p0);
  atomicAdd(&s_load[lane + 32], p1);
  if (lane == 0) atomicAdd(&s_ent, ent);
  __syncthreads();
  if (tid < EE) atomicAdd(&accum[tid], s_load[tid]);
  if (tid == 0) atomicAdd(&accum[EE], s_ent);
}

__global__ void stats_kernel(const float* __restrict__ accum,
                             float* __restrict__ out_var, float* __restrict__ out_ent) {
  const int lane = threadIdx.x & 31;
  float l0 = accum[lane] / (float)BB;
  float l1 = accum[lane + 32] / (float)BB;
  float s = l0 + l1;
#pragma unroll
  for (int off = 16; off > 0; off >>= 1) s += __shfl_xor(s, off, 32);
  const float mean = s / (float)EE;
  float d = (l0 - mean) * (l0 - mean) + (l1 - mean) * (l1 - mean);
#pragma unroll
  for (int off = 16; off > 0; off >>= 1) d += __shfl_xor(d, off, 32);
  if (lane == 0) {
    *out_var = d / (float)(EE - 1);      // unbiased variance
    *out_ent = accum[EE] / (float)BB;
  }
}

// ---------------- launch ----------------
extern "C" void kernel_launch(void* const* d_in, const int* in_sizes, int n_in,
                              void* d_out, int out_size, void* d_ws, size_t ws_size,
                              hipStream_t stream) {
  (void)in_sizes; (void)n_in; (void)out_size; (void)ws_size;
  const float* hs = (const float*)d_in[0];
  const float* rw = (const float*)d_in[1];
  const float* sm = (const float*)d_in[2];
  const float* w1 = (const float*)d_in[3];
  const float* b1 = (const float*)d_in[4];
  const float* w2 = (const float*)d_in[5];
  const float* b2 = (const float*)d_in[6];

  char* ws = (char*)d_ws;
  __bf16* Xb    = (__bf16*)(ws + OFF_XB);
  __bf16* W1b   = (__bf16*)(ws + OFF_W1B);
  __bf16* W2b   = (__bf16*)(ws + OFF_W2B);
  __bf16* Rb    = (__bf16*)(ws + OFF_RB);
  __bf16* Hb    = (__bf16*)(ws + OFF_HB);
  float*  gates = (float*)(ws + OFF_GATES);
  float*  accum = (float*)(ws + OFF_ACC);

  float* out        = (float*)d_out;
  float* out_scaled = out;
  float* out_idx    = out + (size_t)BB * EE;
  float* out_w      = out_idx + (size_t)BB * TOPK;
  float* out_var    = out_w + (size_t)BB * TOPK;
  float* out_ent    = out_var + 1;

  const size_t nX  = (size_t)BB * HD;
  const size_t nW1 = (size_t)HHD * HD;
  const size_t nW2 = (size_t)EE * HHD;
  const size_t nR  = (size_t)EE * HD;

  f32_to_bf16_kernel<<<(unsigned)(nX  / 4 / 256), 256, 0, stream>>>((const v4f*)hs, (v4bf*)Xb, nX / 4);
  f32_to_bf16_kernel<<<(unsigned)(nW1 / 4 / 256), 256, 0, stream>>>((const v4f*)w1, (v4bf*)W1b, nW1 / 4);
  f32_to_bf16_kernel<<<(unsigned)(nW2 / 4 / 256), 256, 0, stream>>>((const v4f*)w2, (v4bf*)W2b, nW2 / 4);
  mask_router_kernel<<<(unsigned)(nR / 4 / 256), 256, 0, stream>>>((const v4f*)rw, (const v4f*)sm, (v4bf*)Rb, nR / 4);

  gemm1_relu_kernel<<<(BB / 128) * (HHD / 128), 256, 0, stream>>>(Xb, W1b, b1, Hb);
  gemm2_sigmoid_kernel<<<(BB / 16) / 8, 256, 0, stream>>>(Hb, W2b, b2, gates);
  router_kernel<<<(BB / 16) / 8, 256, 0, stream>>>(Xb, Rb, gates, out_scaled);

  zero_accum_kernel<<<1, 128, 0, stream>>>(accum);
  finalize_kernel<<<BB / 8, 256, 0, stream>>>(out_scaled, out_idx, out_w, accum);
  stats_kernel<<<1, 32, 0, stream>>>(accum, out_var, out_ent);
}
